// GuardedLayer_57140244906441
// MI455X (gfx1250) — matrix operand order repaired and checked by reading, not compiled, not run on hardware
//
#include <hip/hip_runtime.h>

// GuardedLayer: out[n] = sum_c (presence[n,c] > EPS) * (x[n] @ W_c + b_c)
// Masked GEMM, K = 8*64 = 512, f16 WMMA w/ f32 accumulate.
// Prep kernel pre-transposes W -> f16 in d_ws; main kernel stages tiles into
// LDS with GLOBAL_LOAD_ASYNC_TO_LDS_B128 (double-buffered W). Bias is applied
// with 4 extra WMMAs (mask x bias), fully unrolled case loop.

#define EPSV    0.0001f
#define DDIM    64
#define CASES   8
#define MTILE   128
#define THREADS 256
#define XROW    68   // f32 x-tile row stride: 272B, 16B aligned, conflict-free b128
#define WROW    72   // f16 W^T row stride: 144B, 16B aligned, conflict-free b128

typedef __attribute__((ext_vector_type(16))) _Float16 v16h;
typedef __attribute__((ext_vector_type(8)))  _Float16 v8h;
typedef __attribute__((ext_vector_type(8)))  float    v8f;
typedef __attribute__((ext_vector_type(4)))  float    v4f;
typedef __attribute__((ext_vector_type(4)))  int      v4i;

#if __has_builtin(__builtin_amdgcn_global_load_async_to_lds_b128)
#define ASYNC_LDS 1
#else
#define ASYNC_LDS 0
#endif

__device__ __forceinline__ void copy16_g2l(const void* g, void* l) {
#if ASYNC_LDS
  __builtin_amdgcn_global_load_async_to_lds_b128(
      (__attribute__((address_space(1))) v4i*)g,
      (__attribute__((address_space(3))) v4i*)l, 0, 0);
#else
  *(v4f*)l = *(const v4f*)g;
#endif
}

__device__ __forceinline__ void copy_wait() {
#if ASYNC_LDS
#if __has_builtin(__builtin_amdgcn_s_wait_asynccnt)
  __builtin_amdgcn_s_wait_asynccnt(0);
#else
  asm volatile("s_wait_asynccnt 0x0" ::: "memory");
#endif
#endif
}

// ---------------- prep: WtAll[c][n][i] = (f16) W[c][i][n]  (64KB, L2-hot) ----
__global__ __launch_bounds__(256)
void prep_w(const float* __restrict__ W, unsigned short* __restrict__ wt_raw) {
  _Float16* WtAll = (_Float16*)wt_raw;
  const int c = blockIdx.x;
  const float*  Wc = W + c * DDIM * DDIM;
  _Float16*     Tc = WtAll + c * DDIM * DDIM;
  const int t  = threadIdx.x;
  const int n0 = (t & 15) * 4;
#pragma unroll
  for (int ib = 0; ib < 4; ++ib) {
    const int i = (t >> 4) + ib * 16;
    v4f v = *(const v4f*)(Wc + i * DDIM + n0);
#pragma unroll
    for (int k = 0; k < 4; ++k) Tc[(n0 + k) * DDIM + i] = (_Float16)v[k];
  }
}

// ---------------- main kernel ----------------
template <int USE_WS>
__global__ __launch_bounds__(THREADS)
void guarded_layer_wmma(const float* __restrict__ x,
                        const float* __restrict__ presence,
                        const float* __restrict__ W,
                        const float* __restrict__ b,
                        const unsigned short* __restrict__ wt_raw,
                        float* __restrict__ out,
                        int nrows) {
  __shared__ __align__(16) float    Xf[MTILE][XROW];      // x tile, f32
  __shared__ __align__(16) _Float16 Wt[2][DDIM][WROW];    // W_c^T f16, dbl-buffered
  __shared__ __align__(16) float    Mf[MTILE][CASES];     // 0/1 masks
  __shared__ __align__(16) float    Bl[CASES][DDIM];      // biases

  const _Float16* WtAll = (const _Float16*)wt_raw;
  const int  tid     = threadIdx.x;
  const long rowBase = (long)blockIdx.x * MTILE;

  // ---- stage x tile (f32, async): 2048 x 16B chunks ----
#pragma unroll
  for (int j = 0; j < 8; ++j) {
    const int  e   = tid + j * THREADS;   // 0..2047
    const int  row = e >> 4, c16 = e & 15;
    const long gr  = rowBase + row;
    if (gr < nrows)
      copy16_g2l(x + gr * DDIM + c16 * 4, &Xf[row][c16 * 4]);
  }

  // ---- stage masks ----
  {
    const int  r  = tid >> 1;
    const int  c0 = (tid & 1) * 4;
    const long gr = rowBase + r;
    const v4f  z4 = {};
    v4f p = (gr < nrows) ? *(const v4f*)(presence + gr * CASES + c0) : z4;
#pragma unroll
    for (int k = 0; k < 4; ++k) Mf[r][c0 + k] = (p[k] > EPSV) ? 1.f : 0.f;
  }
  // ---- stage biases ----
  if (tid < (CASES * DDIM) / 4)
    *(v4f*)(&Bl[0][0] + tid * 4) = *(const v4f*)(b + tid * 4);

  // ---- stage W_0^T into buffer 0 ----
  if constexpr (USE_WS) {
#pragma unroll
    for (int j = 0; j < 2; ++j) {
      const int e = tid + j * THREADS;    // 0..511, 16B chunks of 8KB
      const int n = e >> 3, i8 = e & 7;
      copy16_g2l(WtAll + n * DDIM + i8 * 8, &Wt[0][n][i8 * 8]);
    }
  } else {
    const int n0 = (tid & 15) * 4;
#pragma unroll
    for (int ib = 0; ib < 4; ++ib) {
      const int i = (tid >> 4) + ib * 16;
      v4f v = *(const v4f*)(W + i * DDIM + n0);
#pragma unroll
      for (int k = 0; k < 4; ++k) Wt[0][n0 + k][i] = (_Float16)v[k];
    }
  }

  copy_wait();
  __syncthreads();

  // ---- per-wave setup ----
  const int lane = tid & 31;
  const int wave = tid >> 5;
  const int ln   = lane & 15;
  const int half = lane >> 4;
  const int m0   = wave * 16;
  const int off  = half * 8;   // A-fragment K interleave (ISA 7.12.2)

  // x fragments: convert f32 LDS -> f16 once per wave, reused for all 8 cases
  v16h xf[2];
#pragma unroll
  for (int h = 0; h < 2; ++h) {
    const float* lo = &Xf[m0 + ln][h * 32 + off];
    const float* hi = lo + 16;
    v16h a;
#pragma unroll
    for (int k = 0; k < 8; ++k) { a[k] = (_Float16)lo[k]; a[8 + k] = (_Float16)hi[k]; }
    xf[h] = a;
  }

  // per-case row-mask values for this lane's A rows (row = m0 + ln, both halves)
  float mrowf[CASES];
#pragma unroll
  for (int c = 0; c < CASES; ++c) mrowf[c] = Mf[m0 + ln][c];

  // per-lane bias values (loaded UNCONDITIONALLY -> no exec-mask divergence;
  // lanes >= 16 zero them out with a plain register select below)
  float bvf[4][CASES];
#pragma unroll
  for (int t = 0; t < 4; ++t)
#pragma unroll
    for (int c = 0; c < CASES; ++c) bvf[t][c] = Bl[c][t * 16 + ln];

  // ---- bias via WMMA: acc = Mask(16x8, K zero-padded) x bias(8x64) ----
  // A lanes<16 hold K=0..7 -> mask values; all other K slots are zero.
  // B lanes<16 hold K=0..15 -> bias rows 0..7 then zeros; lanes>=16 zero.
  v8f acc[4];
  {
    const bool lowhalf = (half == 0);
    v16h am = {};
#pragma unroll
    for (int c = 0; c < CASES; ++c) {
      const _Float16 mv = (_Float16)mrowf[c];
      am[c] = lowhalf ? mv : (_Float16)0.f;
    }
    const v8f zacc = {};
#pragma unroll
    for (int t = 0; t < 4; ++t) {
      v16h bb = {};
#pragma unroll
      for (int c = 0; c < CASES; ++c) {
        const _Float16 bvh = (_Float16)bvf[t][c];
        bb[c] = lowhalf ? bvh : (_Float16)0.f;
      }
      acc[t] = __builtin_amdgcn_wmma_f32_16x16x32_f16(
          false, am, false, bb, (short)0, zacc, false, false);
    }
  }

  // ---- main loop: stage W_{c+1} (async) while computing case c ----
#pragma unroll
  for (int c = 0; c < CASES; ++c) {
    const int buf = c & 1;

    if (c + 1 < CASES) {
      if constexpr (USE_WS) {
        const _Float16* src = WtAll + (c + 1) * DDIM * DDIM;
#pragma unroll
        for (int j = 0; j < 2; ++j) {
          const int e = tid + j * THREADS;
          const int n = e >> 3, i8 = e & 7;
          copy16_g2l(src + n * DDIM + i8 * 8, &Wt[buf ^ 1][n][i8 * 8]);
        }
      } else {
        const float* Wc = W + (c + 1) * DDIM * DDIM;
        const int n0 = (tid & 15) * 4;
#pragma unroll
        for (int ib = 0; ib < 4; ++ib) {
          const int i = (tid >> 4) + ib * 16;
          v4f v = *(const v4f*)(Wc + i * DDIM + n0);
#pragma unroll
          for (int k = 0; k < 4; ++k) Wt[buf ^ 1][n0 + k][i] = (_Float16)v[k];
        }
      }
    }

    // masked A fragments (per-lane row mask -> cndmask, EXEC stays all-ones)
    const bool on = mrowf[c] > 0.5f;
    const v16h z  = {};
    const v16h a0 = on ? xf[0] : z;
    const v16h a1 = on ? xf[1] : z;

    // load ALL 8 B fragments first so DS waits can be graduated / overlapped
    v16h bfr[8];
#pragma unroll
    for (int h = 0; h < 2; ++h)
#pragma unroll
      for (int t = 0; t < 4; ++t) {
        const v8h* q = (const v8h*)&Wt[buf][t * 16 + ln][h * 32 + half * 8];
        bfr[h * 4 + t] = __builtin_shufflevector(q[0], q[2],
                                                 0, 1, 2, 3, 4, 5, 6, 7,
                                                 8, 9, 10, 11, 12, 13, 14, 15);
      }
#pragma unroll
    for (int h = 0; h < 2; ++h) {
      const v16h ah = h ? a1 : a0;
#pragma unroll
      for (int t = 0; t < 4; ++t)
        acc[t] = __builtin_amdgcn_wmma_f32_16x16x32_f16(
            false, ah, false, bfr[h * 4 + t], (short)0, acc[t], false, false);
    }

    copy_wait();       // W_{c+1} async copy landed (overlapped with compute)
    __syncthreads();   // cross-wave visibility + buffer reuse safety
  }

  // ---- epilogue ----
#pragma unroll
  for (int r = 0; r < 8; ++r) {
    const long gr = rowBase + m0 + r + 8 * half;
    if (gr < nrows) {
      float* orow = out + gr * DDIM + ln;
#pragma unroll
      for (int t = 0; t < 4; ++t) orow[t * 16] = acc[t][r];
    }
  }
}

extern "C" void kernel_launch(void* const* d_in, const int* in_sizes, int n_in,
                              void* d_out, int out_size, void* d_ws, size_t ws_size,
                              hipStream_t stream) {
  const float* x        = (const float*)d_in[0];
  const float* presence = (const float*)d_in[1];
  const float* W        = (const float*)d_in[2];
  const float* b        = (const float*)d_in[3];
  float*       out      = (float*)d_out;

  const int    nrows  = in_sizes[0] / DDIM;
  const int    grid   = (nrows + MTILE - 1) / MTILE;
  const size_t wbytes = (size_t)CASES * DDIM * DDIM * sizeof(unsigned short);
  unsigned short* wt  = (unsigned short*)d_ws;

  if (ws_size >= wbytes) {
    prep_w<<<CASES, 256, 0, stream>>>(W, wt);
    guarded_layer_wmma<1><<<grid, THREADS, 0, stream>>>(x, presence, W, b, wt,
                                                        out, nrows);
  } else {
    guarded_layer_wmma<0><<<grid, THREADS, 0, stream>>>(x, presence, W, b, wt,
                                                        out, nrows);
  }
}
